// DecoderFusion_41704132445017
// MI455X (gfx1250) — compile-verified, matrix operand order checked by reading, not run on hardware
//
#include <hip/hip_runtime.h>
#include <hip/hip_bf16.h>
#include <math.h>

typedef __attribute__((ext_vector_type(16))) _Float16 v16h;
typedef __attribute__((ext_vector_type(8)))  float    v8f;

#define BATCH 8
#define CH    64
#define HEADS 4
#define HID   256

// ---------------------------------------------------------------------------
// WMMA fragment K-index map (16-bit A 16x32 layout per CDNA5 ISA 7.12.2):
// lane 0-15 hold K 0..7 (vgpr0-3) and 16..23 (vgpr4-7); lanes 16-31 hold +8.
__device__ __forceinline__ int kmap(int lane, int i) {
  return ((lane >> 4) << 3) + ((i >> 3) << 4) + (i & 7);
}

// ---------------------------------------------------------------------------
// Y[b,m,p] = sum_k W[m,k] * X[b,k,p]  (+ RES[b,m,p])   -- 1x1 conv as GEMM
// block = 128 threads (4 waves), 64 pixels per block, 16 pixels per wave.
// M,K compile-time so the K loop fully unrolls into back-to-back WMMAs.
template <int M, int K, bool HAS_RES>
__global__ void k_gemm1x1(const float* __restrict__ Wt,   // [M][K]
                          const float* __restrict__ X,
                          float* __restrict__ Y,
                          const float* __restrict__ RES,
                          int HW, int hwsh,                // HW = 1<<hwsh
                          int xBS, int yBS, int resBS) {
  extern __shared__ _Float16 sm[];
  _Float16* Wh = sm;            // M*K halfs
  _Float16* Xh = sm + M * K;    // K*64 halfs
  const int tid  = threadIdx.x;
  const int lane = tid & 31;
  const int wave = tid >> 5;
  const int n0 = blockIdx.x * 64;
  const int b  = n0 >> hwsh;
  const int p0 = n0 & (HW - 1);

  for (int idx = tid; idx < M * K; idx += 128) Wh[idx] = (_Float16)Wt[idx];
  const float* Xb = X + (size_t)b * xBS + p0;
  __builtin_prefetch(Xb, 0, 0);
  for (int idx = tid; idx < K * 64; idx += 128) {
    int k = idx >> 6, j = idx & 63;
    Xh[idx] = (_Float16)Xb[k * HW + j];
  }
  __syncthreads();

  const int j0  = wave * 16;
  const int col = lane & 15;
  int kidx[16];
#pragma unroll
  for (int i = 0; i < 16; ++i) kidx[i] = kmap(lane, i);

  const int nidx = p0 + j0 + col;
  float* yb = Y + (size_t)b * yBS + nidx;
  const float* rb = HAS_RES ? (RES + (size_t)b * resBS + nidx) : nullptr;
  const _Float16* xcol = Xh + j0 + col;

#pragma unroll
  for (int m0 = 0; m0 < M; m0 += 16) {
    const _Float16* wrow = Wh + (m0 + col) * K;
    v8f acc = {};
#pragma unroll
    for (int k0 = 0; k0 < K; k0 += 32) {
      v16h a, bq;
#pragma unroll
      for (int i = 0; i < 16; ++i) {
        int kk = k0 + kidx[i];
        a[i]  = wrow[kk];
        bq[i] = xcol[kk * 64];
      }
      acc = __builtin_amdgcn_wmma_f32_16x16x32_f16(false, a, false, bq,
                                                   (short)0, acc, false, false);
    }
    const int mb = m0 + ((lane >> 4) << 3);
    if (HAS_RES) {
      float rv[8];
#pragma unroll
      for (int r = 0; r < 8; ++r) rv[r] = rb[(mb + r) * HW];
#pragma unroll
      for (int r = 0; r < 8; ++r) acc[r] += rv[r];
    }
#pragma unroll
    for (int r = 0; r < 8; ++r) yb[(mb + r) * HW] = acc[r];
  }
}

// ---------------------------------------------------------------------------
// 3x3 conv (pad=1) as 9-tap channel GEMM. Input split over two pointers
// (concat): ci < KSPLIT -> X0, else X1 (compile-time per k-tile).
template <int K, int KSPLIT>
__global__ void k_conv3x3(const float* __restrict__ Wt,   // [M][K][3][3]
                          const float* __restrict__ bias,
                          const float* __restrict__ X0,
                          const float* __restrict__ X1,
                          float* __restrict__ Y,
                          int H, int Wd, int wsh, int hwsh,
                          int xBS, int yBS) {
  const int HW   = H * Wd;
  const int tid  = threadIdx.x;
  const int lane = tid & 31;
  const int wave = tid >> 5;
  const int n0   = blockIdx.x * 64;
  const int b    = n0 >> hwsh;
  const int p    = (n0 & (HW - 1)) + wave * 16 + (lane & 15);
  const int h    = p >> wsh, w = p & (Wd - 1);
  const int col  = lane & 15;
  const float* X0b = X0 + (size_t)b * xBS;
  const float* X1b = X1 + (size_t)b * xBS;
  __builtin_prefetch(X0b + p, 0, 0);

  int kidx[16];
#pragma unroll
  for (int i = 0; i < 16; ++i) kidx[i] = kmap(lane, i);

  float* yb = Y + (size_t)b * yBS + p;

  for (int m0 = 0; m0 < 64; m0 += 16) {
    v8f acc = {};
    for (int dy = 0; dy < 3; ++dy) {
      int hh = h + dy - 1;
      for (int dx = 0; dx < 3; ++dx) {
        int ww = w + dx - 1;
        bool inb = (hh >= 0) && (hh < H) && (ww >= 0) && (ww < Wd);
        int off = hh * Wd + ww;
        int tap = dy * 3 + dx;
        const float* wr = Wt + ((m0 + col) * K) * 9 + tap;
#pragma unroll
        for (int k0 = 0; k0 < K; k0 += 32) {
          const float* src = (k0 < KSPLIT) ? X0b : X1b;
          const int csub = (k0 < KSPLIT) ? 0 : KSPLIT;
          v16h a, bq;
#pragma unroll
          for (int i = 0; i < 16; ++i) {
            int ci = k0 + kidx[i];
            a[i] = (_Float16)wr[ci * 9];
            float xv = inb ? src[(ci - csub) * HW + off] : 0.f;
            bq[i] = (_Float16)xv;
          }
          acc = __builtin_amdgcn_wmma_f32_16x16x32_f16(false, a, false, bq,
                                                       (short)0, acc, false, false);
        }
      }
    }
    const int mb = m0 + ((lane >> 4) << 3);
    float bv[8];
#pragma unroll
    for (int r = 0; r < 8; ++r) bv[r] = bias[mb + r];
#pragma unroll
    for (int r = 0; r < 8; ++r) yb[(mb + r) * HW] = acc[r] + bv[r];
  }
}

// ---------------------------------------------------------------------------
// Depthwise 3x3 (pad=1) + optional mask gating. grid = (HW/256, B*64)
__global__ void k_dw3x3(const float* __restrict__ Wt,   // [64][9]
                        const float* __restrict__ X,
                        const float* __restrict__ mask,
                        int maskmode,                    // 0 none, 1 *m, 2 *(1-m)
                        float* __restrict__ Y,
                        int H, int Wd, int wsh, int xBS, int yBS) {
  int HW = H * Wd;
  int p = blockIdx.x * 256 + threadIdx.x;
  if (p >= HW) return;
  int bc = blockIdx.y;
  int b = bc >> 6, c = bc & 63;
  int h = p >> wsh, w = p & (Wd - 1);
  const float* xb = X + (size_t)b * xBS + c * HW;
  float s = 0.f;
  for (int dy = 0; dy < 3; ++dy) {
    int hh = h + dy - 1;
    if (hh < 0 || hh >= H) continue;
    for (int dx = 0; dx < 3; ++dx) {
      int ww = w + dx - 1;
      if (ww < 0 || ww >= Wd) continue;
      s += Wt[c * 9 + dy * 3 + dx] * xb[hh * Wd + ww];
    }
  }
  if (maskmode == 1) s *= mask[b * HW + p];
  else if (maskmode == 2) s *= (1.f - mask[b * HW + p]);
  Y[(size_t)b * yBS + c * HW + p] = s;
}

// mask[b,p] = sigmoid(bias + sum_c w[c]*x[b,c,p]).  grid = (HW/256, B)
__global__ void k_mask(const float* __restrict__ X, const float* __restrict__ w,
                       const float* __restrict__ bptr, float* __restrict__ mask,
                       int HW, int xBS) {
  int p = blockIdx.x * 256 + threadIdx.x;
  if (p >= HW) return;
  int b = blockIdx.y;
  float s = bptr[0];
  const float* xb = X + (size_t)b * xBS + p;
  for (int c = 0; c < CH; ++c) s += w[c] * xb[c * HW];
  mask[b * HW + p] = 1.f / (1.f + expf(-s));
}

// rn[b*64+c] = 1 / max(||X[b,c,:]||_2, 1e-12)
__global__ void k_rownorm(const float* __restrict__ X, float* __restrict__ rn,
                          int HW, int xBS) {
  __shared__ float red[256];
  int row = blockIdx.x;
  int b = row >> 6, c = row & 63;
  const float* xr = X + (size_t)b * xBS + c * HW;
  float s = 0.f;
  for (int p = threadIdx.x; p < HW; p += 256) { float v = xr[p]; s += v * v; }
  red[threadIdx.x] = s; __syncthreads();
  for (int st = 128; st > 0; st >>= 1) {
    if (threadIdx.x < st) red[threadIdx.x] += red[threadIdx.x + st];
    __syncthreads();
  }
  if (threadIdx.x == 0) rn[row] = 1.f / fmaxf(sqrtf(red[0]), 1e-12f);
}

// attn[bh,c,d] = temp[h]*rq[c]*rk[d] * sum_p q[c,p]*k[d,p]  (one wave per b,h)
__global__ void k_attn_qk(const float* __restrict__ Q, const float* __restrict__ Kt,
                          const float* __restrict__ rq, const float* __restrict__ rk,
                          const float* __restrict__ temp, float* __restrict__ attn,
                          int HW, int BS) {
  int bh = blockIdx.x;
  int b = bh >> 2, head = bh & 3;
  int lane = threadIdx.x;
  int col = lane & 15;
  const float* qrow = Q  + (size_t)b * BS + (head * 16 + col) * HW;
  const float* krow = Kt + (size_t)b * BS + (head * 16 + col) * HW;
  int kidx[16];
#pragma unroll
  for (int i = 0; i < 16; ++i) kidx[i] = kmap(lane, i);
  v8f acc = {};
  for (int p0 = 0; p0 < HW; p0 += 32) {
    v16h a, bq;
#pragma unroll
    for (int i = 0; i < 16; ++i) {
      int kk = p0 + kidx[i];
      a[i]  = (_Float16)qrow[kk];
      bq[i] = (_Float16)krow[kk];
    }
    acc = __builtin_amdgcn_wmma_f32_16x16x32_f16(false, a, false, bq,
                                                 (short)0, acc, false, false);
  }
  float t = temp[head];
#pragma unroll
  for (int r = 0; r < 8; ++r) {
    int m = r + ((lane >> 4) << 3);
    attn[bh * 256 + m * 16 + col] = acc[r] * rq[bh * 16 + m] * rk[bh * 16 + col] * t;
  }
}

// softmax over last dim of attn[512][16]
__global__ void k_softmax16(float* __restrict__ attn, int nrows) {
  int r = blockIdx.x * blockDim.x + threadIdx.x;
  if (r >= nrows) return;
  float* row = attn + r * 16;
  float mx = row[0];
#pragma unroll
  for (int i = 1; i < 16; ++i) mx = fmaxf(mx, row[i]);
  float s = 0.f;
#pragma unroll
  for (int i = 0; i < 16; ++i) { float e = expf(row[i] - mx); row[i] = e; s += e; }
  float inv = 1.f / s;
#pragma unroll
  for (int i = 0; i < 16; ++i) row[i] *= inv;
}

// O[b, h*16+c, p] = sum_d attn[bh,c,d] * V[b, h*16+d, p]  (K padded 16->32)
__global__ void k_attn_v(const float* __restrict__ attn, const float* __restrict__ V,
                         float* __restrict__ O, int HW, int vBS, int oBS) {
  int tile = blockIdx.x * 4 + (threadIdx.x >> 5);
  int lane = threadIdx.x & 31;
  int tilesPerBH = HW >> 4;
  int bh = tile / tilesPerBH;
  int p0 = (tile % tilesPerBH) << 4;
  int b = bh >> 2, head = bh & 3;
  int col = lane & 15;
  const float* vb = V + (size_t)b * vBS + head * 16 * HW + p0 + col;
  const float* ar = attn + bh * 256 + col * 16;
  v16h a, bq;
#pragma unroll
  for (int i = 0; i < 16; ++i) {
    int kk = kmap(lane, i);
    a[i]  = (kk < 16) ? (_Float16)ar[kk] : (_Float16)0.f;
    bq[i] = (kk < 16) ? (_Float16)vb[kk * HW] : (_Float16)0.f;
  }
  v8f acc = {};
  acc = __builtin_amdgcn_wmma_f32_16x16x32_f16(false, a, false, bq,
                                               (short)0, acc, false, false);
  float* ob = O + (size_t)b * oBS + (head * 16 + ((lane >> 4) << 3)) * HW + p0 + col;
#pragma unroll
  for (int r = 0; r < 8; ++r) ob[r * HW] = acc[r];
}

// FFN depthwise 3x3 + exact-GELU gate.  grid = (HW/256, B*256)
__global__ void k_dwgate(const float* __restrict__ Y12, const float* __restrict__ Wdw,
                         float* __restrict__ G, int H, int Wd, int wsh) {
  int HW = H * Wd;
  int p = blockIdx.x * 256 + threadIdx.x;
  if (p >= HW) return;
  int bc = blockIdx.y;
  int b = bc >> 8, c = bc & 255;
  int h = p >> wsh, w = p & (Wd - 1);
  const float* x1 = Y12 + (size_t)b * 512 * HW + c * HW;
  const float* x2 = x1 + HID * HW;
  float s1 = 0.f, s2 = 0.f;
  for (int dy = 0; dy < 3; ++dy) {
    int hh = h + dy - 1;
    if (hh < 0 || hh >= H) continue;
    for (int dx = 0; dx < 3; ++dx) {
      int ww = w + dx - 1;
      if (ww < 0 || ww >= Wd) continue;
      int t = dy * 3 + dx, off = hh * Wd + ww;
      s1 += Wdw[c * 9 + t] * x1[off];
      s2 += Wdw[(c + HID) * 9 + t] * x2[off];
    }
  }
  float g = 0.5f * s1 * (1.f + erff(s1 * 0.70710678118f));
  G[(size_t)b * HID * HW + c * HW + p] = g * s2;
}

// O = A * B elementwise.  grid = (HW/256, B*64)
__global__ void k_mul(const float* __restrict__ A, const float* __restrict__ Bp,
                      float* __restrict__ O, int HW, int aBS, int bBS, int oBS) {
  int p = blockIdx.x * 256 + threadIdx.x;
  if (p >= HW) return;
  int bc = blockIdx.y;
  int b = bc >> 6, c = bc & 63;
  O[(size_t)b * oBS + c * HW + p] =
      A[(size_t)b * aBS + c * HW + p] * Bp[(size_t)b * bBS + c * HW + p];
}

// BatchNorm training stats per channel (biased var)
__global__ void k_bnstats(const float* __restrict__ Yv, float* __restrict__ mean,
                          float* __restrict__ rstd, int HW, int yBS) {
  __shared__ float s1[256], s2[256];
  int c = blockIdx.x;
  float a = 0.f, q = 0.f;
  for (int i = threadIdx.x; i < BATCH * HW; i += 256) {
    int b = i / HW, p = i - b * HW;
    float v = Yv[(size_t)b * yBS + c * HW + p];
    a += v; q += v * v;
  }
  s1[threadIdx.x] = a; s2[threadIdx.x] = q; __syncthreads();
  for (int st = 128; st > 0; st >>= 1) {
    if (threadIdx.x < st) { s1[threadIdx.x] += s1[threadIdx.x + st];
                            s2[threadIdx.x] += s2[threadIdx.x + st]; }
    __syncthreads();
  }
  if (threadIdx.x == 0) {
    float inv = 1.f / (float)(BATCH * HW);
    float m = s1[0] * inv;
    float var = s2[0] * inv - m * m;
    mean[c] = m;
    rstd[c] = rsqrtf(var + 1e-5f);
  }
}

// grid = (HW/256, B*64)
__global__ void k_bnrelu(const float* __restrict__ Yv, const float* __restrict__ mean,
                         const float* __restrict__ rstd, const float* __restrict__ g,
                         const float* __restrict__ be, float* __restrict__ O,
                         int HW, int yBS) {
  int p = blockIdx.x * 256 + threadIdx.x;
  if (p >= HW) return;
  int bc = blockIdx.y;
  int b = bc >> 6, c = bc & 63;
  float v = (Yv[(size_t)b * yBS + c * HW + p] - mean[c]) * rstd[c] * g[c] + be[c];
  O[(size_t)b * CH * HW + c * HW + p] = fmaxf(v, 0.f);
}

// O = FIN + bilinear_up2(SRC), half-pixel.  grid = (HW/256, B*64)
__global__ void k_upadd(const float* __restrict__ FIN, const float* __restrict__ SRC,
                        float* __restrict__ O, int H, int Wd, int wsh) {
  int HW = H * Wd;
  int p = blockIdx.x * 256 + threadIdx.x;
  if (p >= HW) return;
  int bc = blockIdx.y;
  int b = bc >> 6, c = bc & 63;
  int h = p >> wsh, w = p & (Wd - 1);
  int Hs = H >> 1, Ws = Wd >> 1;
  float sy = (h + 0.5f) * 0.5f - 0.5f;
  float sx = (w + 0.5f) * 0.5f - 0.5f;
  float y0f = floorf(sy), x0f = floorf(sx);
  float wy = sy - y0f, wx = sx - x0f;
  int y0 = min(Hs - 1, max(0, (int)y0f));
  int y1 = min(Hs - 1, max(0, (int)y0f + 1));
  int x0 = min(Ws - 1, max(0, (int)x0f));
  int x1 = min(Ws - 1, max(0, (int)x0f + 1));
  const float* s = SRC + ((size_t)b * CH + c) * Hs * Ws;
  float v = (1.f - wy) * ((1.f - wx) * s[y0 * Ws + x0] + wx * s[y0 * Ws + x1]) +
            wy * ((1.f - wx) * s[y1 * Ws + x0] + wx * s[y1 * Ws + x1]);
  size_t oi = ((size_t)b * CH + c) * HW + p;
  O[oi] = FIN[oi] + v;
}

// ===========================================================================
// Host orchestration
// ===========================================================================
namespace {

struct HeadP {
  const float *temp, *q0, *k0, *v0, *qd, *kd, *vd, *proj, *ffn_in, *ffn_dw, *ffn_out;
};

struct Ctx {
  float *bx, *bxf, *bxb, *bbig, *by12, *bmask, *brq, *brk, *batt, *bmean, *brstd;
  const float *mask_w, *mask_b, *fuse_w, *fuse_b, *out_w, *out_b, *bn_g, *bn_b;
  HeadP fg, bg;
  hipStream_t st;
};

static inline int cdiv(long a, long b) { return (int)((a + b - 1) / b); }
static inline int ilog2(int v) { int s = 0; while ((1 << s) < v) ++s; return s; }

template <int M, int K, bool HAS_RES>
static void gemm(const Ctx& c, const float* W, const float* X, float* Y,
                 const float* RES, int HW, int xBS, int yBS, int resBS) {
  int blocks = (BATCH * HW) / 64;
  size_t shmem = (size_t)(M * K + K * 64) * sizeof(_Float16);
  k_gemm1x1<M, K, HAS_RES><<<blocks, 128, shmem, c.st>>>(
      W, X, Y, RES, HW, ilog2(HW), xBS, yBS, resBS);
}

static void run_head(const Ctx& c, const HeadP& P, int mm, float* xh,
                     int H, int W) {
  int HW = H * W, wsh = ilog2(W), hwsh = ilog2(HW);
  int BS64 = 64 * HW, BS256 = 256 * HW, BS512 = 512 * HW;
  float* A  = c.bbig;               // tmp / attention output (ch 0-63)
  float* Q  = c.bbig + (size_t)64 * HW;
  float* Kb = c.bbig + (size_t)128 * HW;
  float* V  = c.bbig + (size_t)192 * HW;
  dim3 g64(cdiv(HW, 256), BATCH * CH);

  // q / k / v = dwconv(1x1conv(x)) (* mask gate for q,k)
  gemm<64, 64, false>(c, P.q0, c.bx, A, nullptr, HW, BS64, BS256, 0);
  k_dw3x3<<<g64, 256, 0, c.st>>>(P.qd, A, c.bmask, mm, Q, H, W, wsh, BS256, BS256);
  gemm<64, 64, false>(c, P.k0, c.bx, A, nullptr, HW, BS64, BS256, 0);
  k_dw3x3<<<g64, 256, 0, c.st>>>(P.kd, A, c.bmask, mm, Kb, H, W, wsh, BS256, BS256);
  gemm<64, 64, false>(c, P.v0, c.bx, A, nullptr, HW, BS64, BS256, 0);
  k_dw3x3<<<g64, 256, 0, c.st>>>(P.vd, A, c.bmask, 0, V, H, W, wsh, BS256, BS256);

  // attention (L2 normalization folded into post-scale of q.k^T)
  k_rownorm<<<BATCH * 64, 256, 0, c.st>>>(Q, c.brq, HW, BS256);
  k_rownorm<<<BATCH * 64, 256, 0, c.st>>>(Kb, c.brk, HW, BS256);
  k_attn_qk<<<BATCH * HEADS, 32, 0, c.st>>>(Q, Kb, c.brq, c.brk, P.temp, c.batt, HW, BS256);
  k_softmax16<<<2, 256, 0, c.st>>>(c.batt, BATCH * HEADS * 16);
  k_attn_v<<<(BATCH * HEADS * (HW / 16)) / 4, 128, 0, c.st>>>(c.batt, V, A, HW, BS256, BS256);

  // xh = x + proj(attn_out)
  gemm<64, 64, true>(c, P.proj, A, xh, c.bx, HW, BS256, BS64, BS64);

  // FFN: xh = xh + ffn_out( gelu(dw(ffn_in(xh)[:256])) * dw(ffn_in(xh)[256:]) )
  gemm<512, 64, false>(c, P.ffn_in, xh, c.by12, nullptr, HW, BS64, BS512, 0);
  k_dwgate<<<dim3(cdiv(HW, 256), BATCH * HID), 256, 0, c.st>>>(c.by12, P.ffn_dw,
                                                               c.bbig, H, W, wsh);
  gemm<64, 256, true>(c, P.ffn_out, c.bbig, xh, xh, HW, BS256, BS64, BS64);
}

static void run_msa(const Ctx& c, int H, int W, float* out) {
  int HW = H * W, wsh = ilog2(W), hwsh = ilog2(HW);
  int BS64 = 64 * HW, BS256 = 256 * HW;
  dim3 g64(cdiv(HW, 256), BATCH * CH);

  k_mask<<<dim3(cdiv(HW, 256), BATCH), 256, 0, c.st>>>(c.bx, c.mask_w, c.mask_b,
                                                       c.bmask, HW, BS64);
  run_head(c, c.fg, 1, c.bxf, H, W);   // foreground head, gate = mask
  run_head(c, c.bg, 2, c.bxb, H, W);   // background head, gate = 1 - mask

  float* F    = c.bbig;                       // fuse output (ch 0-63 of big)
  float* PROD = c.bbig + (size_t)64 * HW;
  float* Yv   = c.bbig + (size_t)128 * HW;

  // f = conv3x3(concat([xb, xf]))
  k_conv3x3<128, 64><<<(BATCH * HW) / 64, 128, 0, c.st>>>(
      c.fuse_w, c.fuse_b, c.bxb, c.bxf, F, H, W, wsh, hwsh, BS64, BS256);
  k_mul<<<g64, 256, 0, c.st>>>(c.bx, F, PROD, HW, BS64, BS256, BS256);
  // y = conv3x3(skip * f)
  k_conv3x3<64, 64><<<(BATCH * HW) / 64, 128, 0, c.st>>>(
      c.out_w, c.out_b, PROD, PROD, Yv, H, W, wsh, hwsh, BS256, BS256);
  k_bnstats<<<64, 256, 0, c.st>>>(Yv, c.bmean, c.brstd, HW, BS256);
  k_bnrelu<<<g64, 256, 0, c.st>>>(Yv, c.bmean, c.brstd, c.bn_g, c.bn_b, out, HW, BS256);
}

} // namespace

extern "C" void kernel_launch(void* const* d_in, const int* in_sizes, int n_in,
                              void* d_out, int out_size, void* d_ws, size_t ws_size,
                              hipStream_t stream) {
  if (n_in < 34) return;
  // Inputs flattened in setup_inputs() dict order:
  // 0..3: f1,f2,f3,f4 ; params in insertion order:
  // 4 mask_w, 5 mask_b,
  // fg: 6 temp,7 q0,8 k0,9 v0,10 qd,11 kd,12 vd,13 proj,14 ffn_in,15 ffn_dw,16 ffn_out
  // bg: 17..27 (same order), 28 fuse_w, 29 fuse_b, 30 out_w, 31 out_b, 32 bn_g, 33 bn_b
  const float* f1 = (const float*)d_in[0];
  const float* f2 = (const float*)d_in[1];
  const float* f3 = (const float*)d_in[2];
  const float* f4 = (const float*)d_in[3];
  auto gp = [&](int i) { return (const float*)d_in[i]; };

  Ctx c;
  c.st = stream;
  c.mask_w = gp(4);  c.mask_b = gp(5);
  c.fg = { gp(6), gp(7), gp(8), gp(9), gp(10), gp(11), gp(12), gp(13), gp(14), gp(15), gp(16) };
  c.bg = { gp(17), gp(18), gp(19), gp(20), gp(21), gp(22), gp(23), gp(24), gp(25), gp(26), gp(27) };
  c.fuse_w = gp(28); c.fuse_b = gp(29);
  c.out_w  = gp(30); c.out_b  = gp(31);
  c.bn_g   = gp(32); c.bn_b   = gp(33);

  // Workspace layout (floats); NP = pixels at largest scale.
  const size_t NP = (size_t)BATCH * 128 * 128;   // 131072
  float* ws = (float*)d_ws;
  c.bx   = ws;                  // 64*NP  current scale input x
  c.bxf  = ws + 64 * NP;        // 64*NP
  c.bxb  = ws + 128 * NP;       // 64*NP
  c.bbig = ws + 192 * NP;       // 256*NP  tmp | q | k | v  (also ffn gate / fuse / y)
  c.by12 = ws + 448 * NP;       // 512*NP  ffn hidden
  c.bmask= ws + 960 * NP;       // NP
  c.brq  = ws + 961 * NP;       // 512
  c.brk  = c.brq + 512;         // 512
  c.batt = c.brk + 512;         // 8192
  c.bmean= c.batt + 8192;       // 64
  c.brstd= c.bmean + 64;        // 64

  // d_out regions in return order (f1, f2, f3, f4)
  float* o1 = (float*)d_out;
  float* o2 = o1 + (size_t)BATCH * CH * 128 * 128;
  float* o3 = o2 + (size_t)BATCH * CH * 64 * 64;
  float* o4 = o3 + (size_t)BATCH * CH * 32 * 32;

  // scale 4
  hipMemcpyAsync(c.bx, f4, (size_t)BATCH * CH * 16 * 16 * sizeof(float),
                 hipMemcpyDeviceToDevice, stream);
  run_msa(c, 16, 16, o4);
  // scale 3: x = f3 + up2(f4_out)
  k_upadd<<<dim3(cdiv(32 * 32, 256), BATCH * CH), 256, 0, stream>>>(f3, o4, c.bx, 32, 32, 5);
  run_msa(c, 32, 32, o3);
  // scale 2
  k_upadd<<<dim3(cdiv(64 * 64, 256), BATCH * CH), 256, 0, stream>>>(f2, o3, c.bx, 64, 64, 6);
  run_msa(c, 64, 64, o2);
  // scale 1
  k_upadd<<<dim3(cdiv(128 * 128, 256), BATCH * CH), 256, 0, stream>>>(f1, o2, c.bx, 128, 128, 7);
  run_msa(c, 128, 128, o1);
}
